// PQNgrammer_11192684773822
// MI455X (gfx1250) — matrix operand order repaired and checked by reading, not compiled, not run on hardware
//
#include <hip/hip_runtime.h>
#include <hip/hip_bf16.h>

typedef __attribute__((ext_vector_type(16))) __bf16 v16bf;
typedef __attribute__((ext_vector_type(8)))  float  v8f;

#define NUM_CLUSTERS 256
#define NUM_HEADS    16
#define DIM          64
#define NGRAM_VOCAB  (768*256)
#define E_DIM        8
#define BL           16384          /* B*L */
#define HD           1024           /* H*D */
#define ROWP         72             /* padded LDS row stride (bf16 elems): 144B, 16B-aligned, bank-spread */

// ---------------- init: zero EMA accumulators, compute hash primes ----------
__global__ __launch_bounds__(256) void pqn_init_kernel(float* __restrict__ zero_base,
                                                       int* __restrict__ primes, int nzero) {
    int idx = blockIdx.x * blockDim.x + threadIdx.x;
    if (idx < nzero) zero_base[idx] = 0.f;
    if (blockIdx.x == 0 && threadIdx.x == 0) {
        int n = NGRAM_VOCAB, found = 0;
        while (found < NUM_HEADS) {
            n++;
            bool p = (n & 1);
            for (int q = 3; p && q * q <= n; q += 2)
                if (n % q == 0) p = false;
            if (p) primes[found++] = n;
        }
    }
}

// ---------------- cluster assignment via split-bf16 WMMA --------------------
// grid = (BL/256, H), block = 128 (4 waves); each wave does 4 token-tiles of 16.
__global__ __launch_bounds__(128) void pqn_assign_kernel(
    const float* __restrict__ x, const float* __restrict__ means,
    int* __restrict__ cid)
{
    extern __shared__ char smem_raw[];
    __bf16* sm_hi  = (__bf16*)smem_raw;                       // [256][ROWP]
    __bf16* sm_lo  = sm_hi + NUM_CLUSTERS * ROWP;             // [256][ROWP]
    float*  sm_msq = (float*)(sm_lo + NUM_CLUSTERS * ROWP);   // [256]

    const int tid  = threadIdx.x;
    const int lane = tid & 31;
    const int wave = tid >> 5;
    const int h    = blockIdx.y;
    const float* mh = means + (size_t)h * (NUM_CLUSTERS * DIM);

    // stage this head's means into LDS as bf16 hi/lo split
    for (int i = tid * 4; i < NUM_CLUSTERS * DIM; i += 128 * 4) {
        float4 m4 = *(const float4*)(mh + i);
        int c = i >> 6, d = i & 63;
        int o = c * ROWP + d;
        float v0=m4.x; __bf16 h0=(__bf16)v0; sm_hi[o+0]=h0; sm_lo[o+0]=(__bf16)(v0-(float)h0);
        float v1=m4.y; __bf16 h1=(__bf16)v1; sm_hi[o+1]=h1; sm_lo[o+1]=(__bf16)(v1-(float)h1);
        float v2=m4.z; __bf16 h2=(__bf16)v2; sm_hi[o+2]=h2; sm_lo[o+2]=(__bf16)(v2-(float)h2);
        float v3=m4.w; __bf16 h3=(__bf16)v3; sm_hi[o+3]=h3; sm_lo[o+3]=(__bf16)(v3-(float)h3);
    }
    __syncthreads();
    for (int c = tid; c < NUM_CLUSTERS; c += 128) {
        float s = 0.f;
        for (int d = 0; d < DIM; ++d) {
            float r = (float)sm_hi[c * ROWP + d] + (float)sm_lo[c * ROWP + d];
            s += r * r;
        }
        sm_msq[c] = s;
    }
    __syncthreads();

    const int base8 = (lane >> 4) << 3;   // A operand: lanes 16-31 take K+8 half
    const int col   = lane & 15;          // B/C operand: cluster column within n-tile
    const int bko   = (lane >> 4) << 3;   // B operand: lanes 16-31 take K+8 half

    for (int it = 0; it < 4; ++it) {
        const int tile   = blockIdx.x * 16 + wave * 4 + it;
        const int token0 = tile * 16;
        const float* xrow = x + (size_t)(token0 + col) * HD + h * DIM;
        if (it < 3) __builtin_prefetch(xrow + (size_t)16 * HD, 0, 0);

        // Build A tiles (16x32 bf16 per k-step) in documented CDNA5 layout:
        // lane<16: row M=lane, K = kb+{0..7,16..23}; lane>=16: row M=lane-16, K = kb+{8..15,24..31}
        v16bf a_hi[2], a_lo[2];
        #pragma unroll
        for (int kb2 = 0; kb2 < 2; ++kb2) {
            const float* p = xrow + kb2 * 32 + base8;
            float4 u0 = *(const float4*)(p);
            float4 u1 = *(const float4*)(p + 4);
            float4 u2 = *(const float4*)(p + 16);
            float4 u3 = *(const float4*)(p + 20);
            float vals[16] = {u0.x,u0.y,u0.z,u0.w, u1.x,u1.y,u1.z,u1.w,
                              u2.x,u2.y,u2.z,u2.w, u3.x,u3.y,u3.z,u3.w};
            #pragma unroll
            for (int j = 0; j < 16; ++j) {
                float v = vals[j];
                __bf16 hi = (__bf16)v;
                a_hi[kb2][j] = hi;
                a_lo[kb2][j] = (__bf16)(v - (float)hi);
            }
        }

        float bestv[8]; int besti[8];
        #pragma unroll
        for (int r = 0; r < 8; ++r) { bestv[r] = 3.4e38f; besti[r] = 0; }

        for (int nt = 0; nt < 16; ++nt) {
            const int c = nt * 16 + col;
            v8f acc = {};
            #pragma unroll
            for (int kb2 = 0; kb2 < 2; ++kb2) {
                // B operand: lane holds K = kb+bko+{0..7,16..23} of one cluster row (16B-aligned LDS loads)
                const __bf16* bp = sm_hi + c * ROWP + kb2 * 32 + bko;
                const __bf16* lp = sm_lo + c * ROWP + kb2 * 32 + bko;
                union { uint4 u[2]; v16bf v; } tb, tl;
                tb.u[0] = *(const uint4*)(bp);     tb.u[1] = *(const uint4*)(bp + 16);
                tl.u[0] = *(const uint4*)(lp);     tl.u[1] = *(const uint4*)(lp + 16);
                // split-bf16 f32-accurate product: hi*hi + hi*lo + lo*hi
                acc = __builtin_amdgcn_wmma_f32_16x16x32_bf16(false, a_hi[kb2], false, tb.v, (short)0, acc, false, false);
                acc = __builtin_amdgcn_wmma_f32_16x16x32_bf16(false, a_hi[kb2], false, tl.v, (short)0, acc, false, false);
                acc = __builtin_amdgcn_wmma_f32_16x16x32_bf16(false, a_lo[kb2], false, tb.v, (short)0, acc, false, false);
            }
            const float msq = sm_msq[c];
            #pragma unroll
            for (int r = 0; r < 8; ++r) {
                float dlt = msq - 2.f * acc[r];   // ||x||^2 dropped (row-constant)
                if (dlt < bestv[r]) { bestv[r] = dlt; besti[r] = c; }
            }
        }

        // argmin across the 16 lanes of each half (rows r / r+8), first-index tie-break
        #pragma unroll
        for (int r = 0; r < 8; ++r) {
            #pragma unroll
            for (int off = 1; off < 16; off <<= 1) {
                float ov = __shfl_xor(bestv[r], off);
                int   oi = __shfl_xor(besti[r], off);
                if (ov < bestv[r] || (ov == bestv[r] && oi < besti[r])) {
                    bestv[r] = ov; besti[r] = oi;
                }
            }
        }
        if (col == 0) {
            int rowbase = token0 + base8;  // lane0 -> rows 0..7, lane16 -> rows 8..15
            #pragma unroll
            for (int r = 0; r < 8; ++r)
                cid[(rowbase + r) * NUM_HEADS + h] = besti[r];
        }
    }
}

// ------- fused: LN(x), EMA scatter, ngram hash + gather, LN(y), output ------
// one wave per (token, head); lane handles 2 features of D=64.
__global__ __launch_bounds__(256) void pqn_fuse_kernel(
    const float* __restrict__ x, const float* __restrict__ embed,
    const float* __restrict__ lnx_s, const float* __restrict__ lnx_b,
    const float* __restrict__ lny_s, const float* __restrict__ lny_b,
    const int* __restrict__ cid, const int* __restrict__ primes,
    float* __restrict__ sumx, float* __restrict__ counts,
    float* __restrict__ out)
{
    const int gw   = (blockIdx.x * blockDim.x + threadIdx.x) >> 5;
    const int lane = threadIdx.x & 31;
    const int t = gw >> 4;   // flattened token b*L + l
    const int h = gw & 15;

    const int d0 = lane * 2;
    float2 v = *(const float2*)(x + (size_t)t * HD + h * DIM + d0);

    float s  = v.x + v.y;
    float ss = v.x * v.x + v.y * v.y;
    #pragma unroll
    for (int off = 16; off > 0; off >>= 1) {
        s  += __shfl_xor(s,  off);
        ss += __shfl_xor(ss, off);
    }
    const float mu   = s * (1.f / 64.f);
    const float var  = ss * (1.f / 64.f) - mu * mu;
    const float rstd = rsqrtf(var + 1e-5f);

    if (d0 < 56) {   // out[..., :D-E] = normed_x
        float2 o;
        o.x = (v.x - mu) * rstd * lnx_s[h * DIM + d0]     + lnx_b[h * DIM + d0];
        o.y = (v.y - mu) * rstd * lnx_s[h * DIM + d0 + 1] + lnx_b[h * DIM + d0 + 1];
        *(float2*)(out + (size_t)t * HD + h * DIM + d0) = o;
    }

    const int k = cid[t * NUM_HEADS + h];
    // EMA accumulation (L2-resident ~1MB table)
    float* sx = sumx + ((size_t)(h * NUM_CLUSTERS + k) * DIM + d0);
    atomicAdd(sx,     v.x);
    atomicAdd(sx + 1, v.y);
    if (lane == 0) atomicAdd(counts + h * NUM_CLUSTERS + k, 1.0f);

    // bigram id -> multi-way hash -> embedding gather -> LN over E=8
    const int linseq = t & 4095;                 // L = 4096
    const int prev   = linseq ? cid[(t - 1) * NUM_HEADS + h] : 0;
    const int ng     = k + prev * NUM_CLUSTERS;  // < 65536
    const int hashed = ((ng * (h + 1) + (h + 1)) % primes[h]) % NGRAM_VOCAB;
    const int row    = hashed + NGRAM_VOCAB * h;

    float yv = 0.f;
    if (lane < E_DIM) yv = embed[(size_t)row * E_DIM + lane];
    float sy = yv, ssy = yv * yv;
    #pragma unroll
    for (int off = 4; off > 0; off >>= 1) {
        sy  += __shfl_xor(sy,  off, 8);
        ssy += __shfl_xor(ssy, off, 8);
    }
    const float muy   = sy * (1.f / 8.f);
    const float vary  = ssy * (1.f / 8.f) - muy * muy;
    const float rstdy = rsqrtf(vary + 1e-5f);
    if (lane < E_DIM) {
        float o = (yv - muy) * rstdy * lny_s[h * E_DIM + lane] + lny_b[h * E_DIM + lane];
        out[(size_t)t * HD + h * DIM + 56 + lane] = o;
    }
}

// ---------------- EMA means finalize ----------------------------------------
__global__ __launch_bounds__(256) void pqn_means_kernel(
    const float* __restrict__ means, const float* __restrict__ sumx,
    const float* __restrict__ counts, float* __restrict__ out_means)
{
    int idx = blockIdx.x * blockDim.x + threadIdx.x;  // < H*K*D
    int h = idx >> 14;
    int k = (idx >> 6) & (NUM_CLUSTERS - 1);
    float cnt = counts[h * NUM_CLUSTERS + k];
    float mx  = sumx[idx] / (1e-6f + cnt);
    out_means[idx] = (1.0f - 0.999f) * mx + 0.999f * means[idx];
}

extern "C" void kernel_launch(void* const* d_in, const int* in_sizes, int n_in,
                              void* d_out, int out_size, void* d_ws, size_t ws_size,
                              hipStream_t stream) {
    const float* x     = (const float*)d_in[0];
    const float* means = (const float*)d_in[1];
    const float* embed = (const float*)d_in[2];
    const float* lnx_s = (const float*)d_in[3];
    const float* lnx_b = (const float*)d_in[4];
    const float* lny_s = (const float*)d_in[5];
    const float* lny_b = (const float*)d_in[6];

    float* out       = (float*)d_out;                    // [B,L,H*D]
    float* out_means = out + (size_t)BL * HD;            // [H,K,D]

    // workspace layout
    int*   cid    = (int*)d_ws;                                  // BL*H ints (1MB)
    float* sumx   = (float*)d_ws + (size_t)BL * NUM_HEADS;       // H*K*D floats (1MB)
    float* counts = sumx + NUM_HEADS * NUM_CLUSTERS * DIM;       // H*K floats
    int*   primes = (int*)(counts + NUM_HEADS * NUM_CLUSTERS);   // 16 ints

    const int nzero = NUM_HEADS * NUM_CLUSTERS * DIM + NUM_HEADS * NUM_CLUSTERS;
    pqn_init_kernel<<<(nzero + 255) / 256, 256, 0, stream>>>(sumx, primes, nzero);

    dim3 agrid(BL / 256, NUM_HEADS);
    size_t lds = (size_t)NUM_CLUSTERS * ROWP * 2 * sizeof(__bf16) + NUM_CLUSTERS * sizeof(float);
    pqn_assign_kernel<<<agrid, 128, lds, stream>>>(x, means, cid);

    pqn_fuse_kernel<<<(BL * NUM_HEADS * 32) / 256, 256, 0, stream>>>(
        x, embed, lnx_s, lnx_b, lny_s, lny_b, cid, primes, sumx, counts, out);

    pqn_means_kernel<<<(NUM_HEADS * NUM_CLUSTERS * DIM) / 256, 256, 0, stream>>>(
        means, sumx, counts, out_means);
}
